// BatchTreeEncoder_74414603371108
// MI455X (gfx1250) — matrix operand order repaired and checked by reading, not compiled, not run on hardware
//
#include <hip/hip_runtime.h>

typedef __attribute__((ext_vector_type(16))) _Float16 v16h;
typedef __attribute__((ext_vector_type(8)))  float    v8f;

#define ENC  128   // EMBED_DIM == ENCODE_DIM == 128
#define MAXD 40
#define ROWS_PER_BLOCK 64   // 4 sub-tiles of 16 rows; B fragments reused 4x

// ---------------- kernel: zero the output accumulator ----------------
__global__ void k_zero(float* __restrict__ out, int n) {
  int i = blockIdx.x * blockDim.x + threadIdx.x;
  if (i < n) out[i] = 0.0f;
}

// ---------------- kernel: Wc fp32 -> fp16, pre-swizzled into WMMA B-fragment order ----
// WcB[((kk*8 + w)*32 + lane)*16 + e] = (f16) Wc[(32*kk + 16*(lane/16) + e)*ENC + 16*w + (lane%16)]
// so that in the GEMM each lane's B operand is one contiguous v16h (32B) load.
__global__ void k_prep_wc(const float* __restrict__ Wc, _Float16* __restrict__ WcB) {
  int tid = blockIdx.x * blockDim.x + threadIdx.x;   // 0..16383
  int e    = tid & 15;
  int lane = (tid >> 4) & 31;
  int w    = (tid >> 9) & 7;
  int kk   = tid >> 12;                              // 0..3
  int k = 32 * kk + 16 * (lane >> 4) + e;            // K index 0..127
  int n = 16 * w + (lane & 15);                      // N index 0..127
  WcB[tid] = (_Float16)Wc[k * ENC + n];
}

// ---- shared GEMM body: x = emb[tokens] @ Wc + bc, WMMA f16 -> f32 acc -------------
// One block = 64 token rows (4 sub-tiles). 8 waves; wave w owns column tile [16w,16w+16).
// Each wave holds its 4 B fragments in registers and reuses them for all 4 sub-tiles.
// GUARDED=false: all 64 rows valid (no clamps, no store guards) -> branch-free codegen.
template <bool GUARDED>
__device__ __forceinline__ void gemm_body(
    int t0,
    const int* __restrict__ tokens,
    const float* __restrict__ emb,
    const _Float16* __restrict__ WcB,
    const float* __restrict__ bc,
    float* __restrict__ v, int T)
{
  // A fragments, already in WMMA lane layout: [subtile][kk][lane][e]
  __shared__ __attribute__((aligned(32))) _Float16 Alds[4][4][32][16];

  const int tid = threadIdx.x;

  // --- gather 64 embedding rows, convert to f16, scatter into A-fragment layout ---
  {
    const int r    = tid >> 4;          // row within sub-tile, == WMMA M
    const int cs   = (tid & 15) * 8;    // 8-wide K chunk (8-aligned -> one fragment run)
    // A-layout (contiguous, matches WcB): K = 16*half + e; invert for this chunk:
    const int kk   = cs >> 5;
    const int k32  = cs & 31;
    const int hf   = (k32 >> 4) & 1;
    const int e0   = (k32 & 8) ? 8 : 0;
    const int lane = 16 * hf + r;
    #pragma unroll
    for (int s = 0; s < 4; ++s) {
      int row = t0 + s * 16 + r;
      if (GUARDED) row = row < T ? row : T - 1;   // clamp -> always a valid token
      int tok = tokens[row];
      const float4* src = (const float4*)(emb + (size_t)tok * ENC + cs);
      float4 f0 = src[0], f1 = src[1];
      alignas(16) _Float16 h[8];
      h[0]=(_Float16)f0.x; h[1]=(_Float16)f0.y; h[2]=(_Float16)f0.z; h[3]=(_Float16)f0.w;
      h[4]=(_Float16)f1.x; h[5]=(_Float16)f1.y; h[6]=(_Float16)f1.z; h[7]=(_Float16)f1.w;
      *(uint4*)&Alds[s][kk][lane][e0] = *(const uint4*)h;   // contiguous 16B store
    }
  }
  __syncthreads();

  const int w    = tid >> 5;     // wave id == column tile
  const int L    = tid & 31;     // lane
  const int n0   = w * 16;
  const int n    = L & 15;
  const int half = L >> 4;

  // B fragments for this wave's column tile, kept in registers for all sub-tiles
  const _Float16* bp = WcB + (size_t)(w * 32 + L) * 16;
  v16h b0 = *(const v16h*)(bp + 0 * 256 * 16);
  v16h b1 = *(const v16h*)(bp + 1 * 256 * 16);
  v16h b2 = *(const v16h*)(bp + 2 * 256 * 16);
  v16h b3 = *(const v16h*)(bp + 3 * 256 * 16);

  const float bias = bc[n0 + n];

  #pragma unroll
  for (int s = 0; s < 4; ++s) {
    v8f c;
    #pragma unroll
    for (int i = 0; i < 8; ++i) c[i] = bias;       // fold bias into accumulator

    v16h a0 = *(const v16h*)&Alds[s][0][L][0];
    c = __builtin_amdgcn_wmma_f32_16x16x32_f16(false, a0, false, b0, (short)0, c, false, false);
    v16h a1 = *(const v16h*)&Alds[s][1][L][0];
    c = __builtin_amdgcn_wmma_f32_16x16x32_f16(false, a1, false, b1, (short)0, c, false, false);
    v16h a2 = *(const v16h*)&Alds[s][2][L][0];
    c = __builtin_amdgcn_wmma_f32_16x16x32_f16(false, a2, false, b2, (short)0, c, false, false);
    v16h a3 = *(const v16h*)&Alds[s][3][L][0];
    c = __builtin_amdgcn_wmma_f32_16x16x32_f16(false, a3, false, b3, (short)0, c, false, false);

    // D layout: c[i] -> row M = i + 8*half, col n0+n. One base pointer, 8 stores
    // at immediate offsets i*ENC*4 = i*512 bytes.
    const int rbase = t0 + s * 16 + 8 * half;
    float* dst = v + (size_t)rbase * ENC + n0 + n;
    if (!GUARDED) {
      #pragma unroll
      for (int i = 0; i < 8; ++i) dst[(size_t)i * ENC] = c[i];
    } else {
      #pragma unroll
      for (int i = 0; i < 8; ++i)
        if (rbase + i < T) dst[(size_t)i * ENC] = c[i];
    }
  }
}

// Full tiles only: launched for the first (T/64)*64 rows. Branch-free.
__global__ void __launch_bounds__(256)
k_gemm_full(const int* __restrict__ tokens, const float* __restrict__ emb,
            const _Float16* __restrict__ WcB, const float* __restrict__ bc,
            float* __restrict__ v, int T)
{
  gemm_body<false>(blockIdx.x * ROWS_PER_BLOCK, tokens, emb, WcB, bc, v, T);
}

// Tail tile (rows [t0, T), < 64 rows): only enqueued when T % 64 != 0.
__global__ void __launch_bounds__(256)
k_gemm_tail(const int* __restrict__ tokens, const float* __restrict__ emb,
            const _Float16* __restrict__ WcB, const float* __restrict__ bc,
            float* __restrict__ v, int T, int t0)
{
  gemm_body<true>(t0, tokens, emb, WcB, bc, v, T);
}

// ---------------- kernel: one depth level of bottom-up tree aggregation ----------------
// 32 threads per node, 4 floats per thread; children add into parents with f32 atomics.
__global__ void k_tree_pass(const int* __restrict__ depth,
                            const int* __restrict__ parent,
                            float* __restrict__ v, int T, int d)
{
  long gid = (long)blockIdx.x * blockDim.x + threadIdx.x;
  int node = (int)(gid >> 5);
  if (node >= T) return;
  if (depth[node] != d) return;
  int p = parent[node];
  if (p >= T) return;                       // dummy root parent (index T)
  int j = ((int)gid & 31) * 4;
  const float4 val = *(const float4*)&v[(size_t)node * ENC + j];
  float* dst = &v[(size_t)p * ENC + j];
  atomicAdd(dst + 0, val.x);
  atomicAdd(dst + 1, val.y);
  atomicAdd(dst + 2, val.z);
  atomicAdd(dst + 3, val.w);
}

// ---------------- kernel: per-batch segment max + ReLU ----------------
// Result is max(0, max over nodes). Accumulator starts at 0.0f (bits 0); values are
// clamped to >=0 before the atomic, so uint compare == float compare.
__global__ void k_segmax(const int* __restrict__ node2batch,
                         const float* __restrict__ v,
                         float* __restrict__ out, int T)
{
  long gid = (long)blockIdx.x * blockDim.x + threadIdx.x;
  int node = (int)(gid >> 5);
  if (node >= T) return;
  int j = ((int)gid & 31) * 4;
  int b = node2batch[node];
  const float4 val = *(const float4*)&v[(size_t)node * ENC + j];
  unsigned* dst = (unsigned*)&out[(size_t)b * ENC + j];
  atomicMax(dst + 0, __float_as_uint(fmaxf(val.x, 0.0f)));
  atomicMax(dst + 1, __float_as_uint(fmaxf(val.y, 0.0f)));
  atomicMax(dst + 2, __float_as_uint(fmaxf(val.z, 0.0f)));
  atomicMax(dst + 3, __float_as_uint(fmaxf(val.w, 0.0f)));
}

extern "C" void kernel_launch(void* const* d_in, const int* in_sizes, int n_in,
                              void* d_out, int out_size, void* d_ws, size_t ws_size,
                              hipStream_t stream) {
  const int*   tokens = (const int*)d_in[0];
  const int*   parent = (const int*)d_in[1];
  const int*   depth  = (const int*)d_in[2];
  const int*   n2b    = (const int*)d_in[3];
  const float* emb    = (const float*)d_in[4];
  const float* Wc     = (const float*)d_in[5];
  const float* bc     = (const float*)d_in[6];
  float* out = (float*)d_out;

  const int T = in_sizes[0];                 // 256000

  // workspace: [ WcB : 128*128 f16 = 32 KB ][ v : T*128 f32 ]
  _Float16* WcB = (_Float16*)d_ws;
  float*    v   = (float*)((char*)d_ws + (size_t)ENC * ENC * sizeof(_Float16));

  k_zero<<<(out_size + 255) / 256, 256, 0, stream>>>(out, out_size);
  k_prep_wc<<<64, 256, 0, stream>>>(Wc, WcB);

  const int full_blocks = T / ROWS_PER_BLOCK;          // 4000 for T=256000
  if (full_blocks > 0)
    k_gemm_full<<<full_blocks, 256, 0, stream>>>(tokens, emb, WcB, bc, v, T);
  if (T % ROWS_PER_BLOCK != 0)                         // never taken for T=256000
    k_gemm_tail<<<1, 256, 0, stream>>>(tokens, emb, WcB, bc, v, T,
                                       full_blocks * ROWS_PER_BLOCK);

  long tot  = (long)T * 32;
  int  blks = (int)((tot + 255) / 256);
  for (int d = MAXD; d >= 1; --d)
    k_tree_pass<<<blks, 256, 0, stream>>>(depth, parent, v, T, d);

  k_segmax<<<blks, 256, 0, stream>>>(n2b, v, out, T);
}
